// GNNEncoder_39737037423417
// MI455X (gfx1250) — compile-verified, matrix operand order checked by reading, not run on hardware
//
#include <hip/hip_runtime.h>
#include <hip/hip_fp16.h>

#define HIDDEN    64
#define EDGE_DIM  17
#define NODE_DIM  25
#define N_GRAPHS  128

typedef __attribute__((ext_vector_type(16))) _Float16 v16h;
typedef __attribute__((ext_vector_type(8)))  _Float16 v8h;
typedef __attribute__((ext_vector_type(8)))  float    v8f;

union V16U { v16h v; v8h h[2]; };

__device__ __forceinline__ v16h cat16(v8h lo, v8h hi) {
  V16U u; u.h[0] = lo; u.h[1] = hi; return u.v;
}

// 16-bit A/B fragment K index for half i (0..15) in a lane (ISA 7.12.2):
// halves 0..7 -> K = c..c+7, halves 8..15 -> K = c+16..c+23, c = (lane&16)?8:0
__device__ __forceinline__ int kmap(int lane, int i) {
  return ((i & 8) << 1) + ((lane & 16) >> 1) + (i & 7);
}

__device__ __forceinline__ v16h ldsFrag(const _Float16* p) {
  return cat16(*(const v8h*)p, *(const v8h*)(p + 8));
}

__device__ __forceinline__ v8f wmma_f16(v16h a, v16h b, v8f c) {
  // D = A(16x32 f16) x B(32x16 f16) + C(16x16 f32)
  return __builtin_amdgcn_wmma_f32_16x16x32_f16(false, a, false, b, (short)0, c,
                                                false, false);
}

__device__ __forceinline__ void atomAddF(float* p, float v) {
  unsafeAtomicAdd(p, v);   // lowers to global_atomic_add_f32 (no return)
}

// Pre-swizzle a [Kdim x 64] f32 weight matrix into per-lane WMMA B fragments in
// LDS. Fragment f = kt*4+nt; per lane 16 contiguous halves. nK = #K-steps of 32.
__device__ __forceinline__ void fillFrags(_Float16* dst, const float* w, int Kdim,
                                          int nK, int tid, int nthreads) {
  const int total = nK * 4 * 512;
  for (int t = tid; t < total; t += nthreads) {
    const int f  = t >> 9;
    const int kt = f >> 2, nt = f & 3;
    const int l  = (t >> 4) & 31, i = t & 15;
    const int K  = kt * 32 + kmap(l, i);
    const int Nc = nt * 16 + (l & 15);
    dst[t] = (_Float16)((K < Kdim) ? w[K * 64 + Nc] : 0.f);
  }
}

// ---------------------------------------------------------------------------
// h0 = relu(x @ win + bin)   (tiny: 0.32 GFLOP total, scalar is fine)
__global__ void __launch_bounds__(256)
proj_kernel(const float* __restrict__ x, const float* __restrict__ win,
            const float* __restrict__ bin_, float* __restrict__ h, int N) {
  const int t = blockIdx.x * 256 + threadIdx.x;
  if (t >= N * HIDDEN) return;
  const int node = t >> 6, col = t & 63;
  float acc = bin_[col];
  const float* xr = x + (size_t)node * NODE_DIM;
#pragma unroll
  for (int k = 0; k < NODE_DIM; ++k) acc = fmaf(xr[k], win[k * 64 + col], acc);
  h[t] = fmaxf(acc, 0.f);
}

__global__ void __launch_bounds__(256)
zero_kernel(float* __restrict__ p, int n) {
  const int t = blockIdx.x * 256 + threadIdx.x;
  if (t < n) p[t] = 0.f;
}

// ---------------------------------------------------------------------------
// One wave32 per 16-edge tile:
//   t   = relu(edge_attr(16x17) @ ew1 + eb1)      4 WMMA (K padded to 32)
//   w   = t(16x64) @ ew2 + eb2                    8 WMMA (2 K-steps x 4 N-tiles)
//   agg[dst] += w * h[src]                        64 f32 atomics / edge row set
__global__ void __launch_bounds__(256)
edge_conv_kernel(const float* __restrict__ h, const int* __restrict__ edge_index,
                 const float* __restrict__ edge_attr,
                 const float* __restrict__ ew1, const float* __restrict__ eb1,
                 const float* __restrict__ ew2, const float* __restrict__ eb2,
                 float* __restrict__ agg, int E) {
  __shared__ __align__(16) _Float16 sB1[4 * 512];   // ew1 fragments (K pad 32)
  __shared__ __align__(16) _Float16 sB2[8 * 512];   // ew2 fragments (K = 64)
  __shared__ __align__(16) _Float16 sT[8][16 * 64]; // per-wave t staging

  const int tid = threadIdx.x;
  fillFrags(sB1, ew1, EDGE_DIM, 1, tid, 256);
  fillFrags(sB2, ew2, HIDDEN, 2, tid, 256);
  __syncthreads();

  const int lane   = tid & 31;
  const int wid    = tid >> 5;
  const int lanelo = lane & 15;
  const int mbase  = (lane & 16) ? 8 : 0;
  const int e0     = (blockIdx.x * 8 + wid) * 16;
  const int eRow   = (e0 + lanelo < E) ? (e0 + lanelo) : (E - 1); // clamp (E%16==0 normally)

  // ---- GEMM1: build A directly from global (16 scalar f32 loads, K<17 live)
  v16h a1;
  {
    const float* row = edge_attr + (size_t)eRow * EDGE_DIM;
#pragma unroll
    for (int i = 0; i < 16; ++i) {
      const int K = kmap(lane, i);
      a1[i] = (_Float16)((K < EDGE_DIM) ? row[K] : 0.f);
    }
  }
  v8f c1[4];
#pragma unroll
  for (int nt = 0; nt < 4; ++nt) {
    v8f z = {};
    c1[nt] = wmma_f16(a1, ldsFrag(&sB1[(nt * 32 + lane) * 16]), z);
  }

  // ---- bias + relu, C-layout -> A-layout transpose through LDS
  _Float16* tbuf = sT[wid];
#pragma unroll
  for (int nt = 0; nt < 4; ++nt) {
    const int col = nt * 16 + lanelo;
    const float b = eb1[col];
#pragma unroll
    for (int v = 0; v < 8; ++v)
      tbuf[(mbase + v) * 64 + col] = (_Float16)fmaxf(c1[nt][v] + b, 0.f);
  }
  __syncthreads();

  // ---- GEMM2
  v8f c2[4];
#pragma unroll
  for (int nt = 0; nt < 4; ++nt) { v8f z = {}; c2[nt] = z; }
#pragma unroll
  for (int kt = 0; kt < 2; ++kt) {
    const int cb = kt * 32 + ((lane & 16) ? 8 : 0);
    const v16h a2 = cat16(*(const v8h*)&tbuf[lanelo * 64 + cb],
                          *(const v8h*)&tbuf[lanelo * 64 + cb + 16]);
#pragma unroll
    for (int nt = 0; nt < 4; ++nt)
      c2[nt] = wmma_f16(a2, ldsFrag(&sB2[((kt * 4 + nt) * 32 + lane) * 16]), c2[nt]);
  }

  // ---- gather-multiply + scatter-add
  int srcv[8], dstv[8];
#pragma unroll
  for (int v = 0; v < 8; ++v) {
    const int e  = e0 + mbase + v;
    const int ec = (e < E) ? e : (E - 1);
    srcv[v] = edge_index[ec];
    dstv[v] = edge_index[E + ec];
  }
#pragma unroll
  for (int nt = 0; nt < 4; ++nt) {
    const int col = nt * 16 + lanelo;
    const float b = eb2[col];
#pragma unroll
    for (int v = 0; v < 8; ++v) {
      if (e0 + mbase + v < E) {
        const float w  = c2[nt][v] + b;
        const float hv = h[(size_t)srcv[v] * 64 + col];
        atomAddF(&agg[(size_t)dstv[v] * 64 + col], w * hv);
      }
    }
  }
}

// ---------------------------------------------------------------------------
// One wave32 per 16-node tile: h' = relu(h@sw + agg@nw + sb + nb), 16 WMMA/tile
__global__ void __launch_bounds__(256)
node_update_kernel(const float* __restrict__ h, const float* __restrict__ agg,
                   const float* __restrict__ sw_, const float* __restrict__ sb,
                   const float* __restrict__ nw_, const float* __restrict__ nb,
                   float* __restrict__ hout, int N) {
  __shared__ __align__(16) _Float16 sSW[8 * 512];
  __shared__ __align__(16) _Float16 sNW[8 * 512];
  const int tid = threadIdx.x;
  fillFrags(sSW, sw_, HIDDEN, 2, tid, 256);
  fillFrags(sNW, nw_, HIDDEN, 2, tid, 256);
  __syncthreads();

  const int lane   = tid & 31;
  const int lanelo = lane & 15;
  const int mbase  = (lane & 16) ? 8 : 0;
  const int n0     = (blockIdx.x * 8 + (tid >> 5)) * 16;
  if (n0 >= N) return;   // whole wave exits; no barriers past this point

  v8f acc[4];
#pragma unroll
  for (int nt = 0; nt < 4; ++nt) { v8f z = {}; acc[nt] = z; }

#pragma unroll
  for (int kt = 0; kt < 2; ++kt) {
    const float* p = h + (size_t)(n0 + lanelo) * 64 + kt * 32 + ((lane & 16) ? 8 : 0);
    v16h a;
#pragma unroll
    for (int i = 0; i < 8; ++i) a[i] = (_Float16)p[i];
#pragma unroll
    for (int i = 0; i < 8; ++i) a[8 + i] = (_Float16)p[16 + i];
#pragma unroll
    for (int nt = 0; nt < 4; ++nt)
      acc[nt] = wmma_f16(a, ldsFrag(&sSW[((kt * 4 + nt) * 32 + lane) * 16]), acc[nt]);
  }
#pragma unroll
  for (int kt = 0; kt < 2; ++kt) {
    const float* p = agg + (size_t)(n0 + lanelo) * 64 + kt * 32 + ((lane & 16) ? 8 : 0);
    v16h a;
#pragma unroll
    for (int i = 0; i < 8; ++i) a[i] = (_Float16)p[i];
#pragma unroll
    for (int i = 0; i < 8; ++i) a[8 + i] = (_Float16)p[16 + i];
#pragma unroll
    for (int nt = 0; nt < 4; ++nt)
      acc[nt] = wmma_f16(a, ldsFrag(&sNW[((kt * 4 + nt) * 32 + lane) * 16]), acc[nt]);
  }

#pragma unroll
  for (int nt = 0; nt < 4; ++nt) {
    const int col = nt * 16 + lanelo;
    const float b = sb[col] + nb[col];
#pragma unroll
    for (int v = 0; v < 8; ++v)
      hout[(size_t)(n0 + mbase + v) * 64 + col] = fmaxf(acc[nt][v] + b, 0.f);
  }
}

// ---------------------------------------------------------------------------
__global__ void __launch_bounds__(256)
pool_kernel(const float* __restrict__ h, const int* __restrict__ batch,
            float* __restrict__ pooled, float* __restrict__ cnt, int N) {
  const int t = blockIdx.x * 256 + threadIdx.x;
  if (t >= N * HIDDEN) return;
  const int node = t >> 6, col = t & 63;
  const int g = batch[node];
  atomAddF(&pooled[g * 64 + col], h[t]);
  if (col == 0) atomAddF(&cnt[g], 1.f);
}

__global__ void __launch_bounds__(256)
fin_kernel(const float* __restrict__ pooled, const float* __restrict__ cnt,
           float* __restrict__ out) {
  const int t = blockIdx.x * 256 + threadIdx.x;
  if (t >= N_GRAPHS * HIDDEN) return;
  out[t] = pooled[t] / fmaxf(cnt[t >> 6], 1.f);
}

// ---------------------------------------------------------------------------
extern "C" void kernel_launch(void* const* d_in, const int* in_sizes, int n_in,
                              void* d_out, int out_size, void* d_ws, size_t ws_size,
                              hipStream_t stream) {
  const float* x          = (const float*)d_in[0];
  const int*   edge_index = (const int*)d_in[1];
  const float* edge_attr  = (const float*)d_in[2];
  const int*   batch      = (const int*)d_in[3];
  const float* win        = (const float*)d_in[4];
  const float* bin_       = (const float*)d_in[5];

  const int N = in_sizes[0] / NODE_DIM;   // 100000
  const int E = in_sizes[2] / EDGE_DIM;   // 1600000

  float* ws     = (float*)d_ws;
  float* hA     = ws;
  float* hB     = hA + (size_t)N * HIDDEN;
  float* agg    = hB + (size_t)N * HIDDEN;
  float* pooled = agg + (size_t)N * HIDDEN;
  float* cnt    = pooled + N_GRAPHS * HIDDEN;

  proj_kernel<<<(N * HIDDEN + 255) / 256, 256, 0, stream>>>(x, win, bin_, hA, N);

  for (int l = 0; l < 3; ++l) {
    const float* ew1 = (const float*)d_in[6 + l * 8 + 0];
    const float* eb1 = (const float*)d_in[6 + l * 8 + 1];
    const float* ew2 = (const float*)d_in[6 + l * 8 + 2];
    const float* eb2 = (const float*)d_in[6 + l * 8 + 3];
    const float* sw_ = (const float*)d_in[6 + l * 8 + 4];
    const float* sb  = (const float*)d_in[6 + l * 8 + 5];
    const float* nw_ = (const float*)d_in[6 + l * 8 + 6];
    const float* nb  = (const float*)d_in[6 + l * 8 + 7];

    zero_kernel<<<(N * HIDDEN + 255) / 256, 256, 0, stream>>>(agg, N * HIDDEN);
    // 128 edges per 256-thread block (8 waves x 16-edge tiles); E % 128 == 0
    edge_conv_kernel<<<(E + 127) / 128, 256, 0, stream>>>(
        hA, edge_index, edge_attr, ew1, eb1, ew2, eb2, agg, E);
    node_update_kernel<<<(N / 16 + 7) / 8, 256, 0, stream>>>(
        hA, agg, sw_, sb, nw_, nb, hB, N);
    float* tmp = hA; hA = hB; hB = tmp;
  }

  zero_kernel<<<(N_GRAPHS * HIDDEN + N_GRAPHS + 255) / 256, 256, 0, stream>>>(
      pooled, N_GRAPHS * HIDDEN + N_GRAPHS);
  pool_kernel<<<(N * HIDDEN + 255) / 256, 256, 0, stream>>>(hA, batch, pooled, cnt, N);
  fin_kernel<<<(N_GRAPHS * HIDDEN + 255) / 256, 256, 0, stream>>>(pooled, cnt,
                                                                  (float*)d_out);
}
